// Transformer_head_46076409152062
// MI455X (gfx1250) — compile-verified, matrix operand order checked by reading, not run on hardware
//
#include <hip/hip_runtime.h>
#include <hip/hip_bf16.h>

// ---------------------------------------------------------------------------
// MI455X (gfx1250) transformer encoder layer, wave32 + WMMA f16->f32.
// ---------------------------------------------------------------------------

typedef __attribute__((ext_vector_type(16))) _Float16 v16h;
typedef __attribute__((ext_vector_type(8)))  float    v8f;
typedef __attribute__((ext_vector_type(4)))  unsigned int v4u;

union AFrag { v16h h; v4u u[2]; };

#define NH_ 12
#define DH_ 64
#define H_  768
#define FF_ 3072
#define S_  1024
#define B_  16

// ---------------------------------------------------------------------------
// DPP cross-lane reductions over 16-lane halves (pure VALU, no LDS traffic)
// ctrl: 0xB1 = quad_perm [1,0,3,2] (xor1), 0x4E = quad_perm [2,3,0,1] (xor2),
//       0x141 = ROW_HALF_MIRROR (xor4-equivalent), 0x140 = ROW_MIRROR (xor8-eq)
// ---------------------------------------------------------------------------
template <int CTRL>
__device__ inline float dppmov(float x) {
  return __builtin_bit_cast(
      float, __builtin_amdgcn_update_dpp(0, __builtin_bit_cast(int, x), CTRL,
                                         0xf, 0xf, true));
}
__device__ inline float rmax16(float x) {
  x = fmaxf(x, dppmov<0xB1>(x));
  x = fmaxf(x, dppmov<0x4E>(x));
  x = fmaxf(x, dppmov<0x141>(x));
  x = fmaxf(x, dppmov<0x140>(x));
  return x;
}
__device__ inline float radd16(float x) {
  x += dppmov<0xB1>(x);
  x += dppmov<0x4E>(x);
  x += dppmov<0x141>(x);
  x += dppmov<0x140>(x);
  return x;
}

// ---------------------------------------------------------------------------
// f32 -> f16 elementwise cast
// ---------------------------------------------------------------------------
__global__ __launch_bounds__(256) void cast_f16_kernel(const float* __restrict__ in,
                                                       _Float16* __restrict__ out, int n) {
  int i = blockIdx.x * 256 + threadIdx.x;
  if (i < n) out[i] = (_Float16)in[i];
}

// ---------------------------------------------------------------------------
// Transpose W[K][N] f32 -> Wt[N][K] f16  (K,N multiples of 32)
// ---------------------------------------------------------------------------
__global__ __launch_bounds__(256) void transpose_f16_kernel(const float* __restrict__ W,
                                                            _Float16* __restrict__ Wt,
                                                            int K, int N) {
  __shared__ float tile[32][33];
  int bx = blockIdx.x;  // N/32
  int by = blockIdx.y;  // K/32
  int tx = threadIdx.x; // 0..31
  int ty = threadIdx.y; // 0..7
  for (int j = 0; j < 4; ++j) {
    int k = by * 32 + ty + j * 8;
    tile[ty + j * 8][tx] = W[(size_t)k * N + bx * 32 + tx];
  }
  __syncthreads();
  for (int j = 0; j < 4; ++j) {
    int n = bx * 32 + ty + j * 8;
    Wt[(size_t)n * K + by * 32 + tx] = (_Float16)tile[tx][ty + j * 8];
  }
}

// ---------------------------------------------------------------------------
// WMMA GEMM: C = A[M][K](f16) x Bt[N][K](f16)^T + bias, templated epilogue.
// Block tile 128x128x32, 256 threads = 8 waves (2 M x 4 N), wave tile 64x32.
// Register-staged LDS double buffering (one barrier per k-step) + explicit
// global_prefetch of tile k+2 to cover HBM latency beyond the register stage.
// EPI: 1 = GELU -> f16 [M][N]
//      2 = f16 -> [B,NH,S,DH]   (Q, K)
//      3 = f16 -> [B,NH,DH,S]   (V transposed)
//      4 = f32 -> v + bias + res[M][N]
// ---------------------------------------------------------------------------
template <int EPI>
__global__ __launch_bounds__(256) void gemm_wmma_kernel(
    const _Float16* __restrict__ A, const _Float16* __restrict__ Bt,
    const float* __restrict__ bias, int M, int N, int K,
    _Float16* __restrict__ outH, float* __restrict__ outF,
    const float* __restrict__ res) {
  const int LDT = 48;  // f16 elements per LDS row (96B, keeps b128 frags aligned)
  __shared__ _Float16 As[2][128 * LDT];
  __shared__ _Float16 Bs[2][128 * LDT];

  int tid = threadIdx.x;
  int w = tid >> 5, lane = tid & 31, lr = lane & 15;
  bool lo = lane < 16;
  int byteSel = lo ? 0 : 16;
  int waveN = w & 3, waveM = w >> 2;
  int rowBase = blockIdx.y * 128;
  int colBase = blockIdx.x * 128;

  v8f zero = {0.f, 0.f, 0.f, 0.f, 0.f, 0.f, 0.f, 0.f};
  v8f acc[4][2];
  for (int mt = 0; mt < 4; ++mt)
    for (int nt = 0; nt < 2; ++nt) acc[mt][nt] = zero;

  // Staging map: 512 16B-chunks (128 rows x 4), 2 per thread.
  int r0 = tid >> 2, c0 = (tid & 3) * 8;
  int r1 = (tid + 256) >> 2, c1 = ((tid + 256) & 3) * 8;
  const _Float16* pA0 = A + (size_t)(rowBase + r0) * K + c0;
  const _Float16* pA1 = A + (size_t)(rowBase + r1) * K + c1;
  const _Float16* pB0 = Bt + (size_t)(colBase + r0) * K + c0;
  const _Float16* pB1 = Bt + (size_t)(colBase + r1) * K + c1;
  bool okA0 = (rowBase + r0) < M, okA1 = (rowBase + r1) < M;

  v4u zu = {0, 0, 0, 0};
  v4u a0 = zu, a1 = zu, b0 = zu, b1 = zu;
  if (okA0) a0 = *(const v4u*)pA0;
  if (okA1) a1 = *(const v4u*)pA1;
  b0 = *(const v4u*)pB0;
  b1 = *(const v4u*)pB1;
  *(v4u*)&As[0][r0 * LDT + c0] = a0;
  *(v4u*)&As[0][r1 * LDT + c1] = a1;
  *(v4u*)&Bs[0][r0 * LDT + c0] = b0;
  *(v4u*)&Bs[0][r1 * LDT + c1] = b1;
  __syncthreads();

  int nk = K >> 5;
#pragma unroll 2
  for (int kt = 0; kt < nk; ++kt) {
    int cur = kt & 1;
    // Issue next tile's global loads early (overlap with WMMAs below).
    if (kt + 1 < nk) {
      int ko = (kt + 1) * 32;
      a0 = zu; a1 = zu;
      if (okA0) a0 = *(const v4u*)(pA0 + ko);
      if (okA1) a1 = *(const v4u*)(pA1 + ko);
      b0 = *(const v4u*)(pB0 + ko);
      b1 = *(const v4u*)(pB1 + ko);
    }
    // L2 prefetch two tiles ahead (global_prefetch_b8: no counters, no VGPRs).
    if (kt + 2 < nk) {
      int ko2 = (kt + 2) * 32;
      if (okA0) __builtin_prefetch(pA0 + ko2, 0, 0);
      __builtin_prefetch(pB0 + ko2, 0, 0);
    }

    AFrag Af[4], Bf[2];
    for (int mt = 0; mt < 4; ++mt) {
      const char* p = (const char*)&As[cur][(waveM * 64 + mt * 16 + lr) * LDT];
      Af[mt].u[0] = *(const v4u*)(p + byteSel);
      Af[mt].u[1] = *(const v4u*)(p + 32 + byteSel);
    }
    for (int nt = 0; nt < 2; ++nt) {
      const char* p = (const char*)&Bs[cur][(waveN * 32 + nt * 16 + lr) * LDT];
      Bf[nt].u[0] = *(const v4u*)(p + byteSel);
      Bf[nt].u[1] = *(const v4u*)(p + 32 + byteSel);
    }
    for (int mt = 0; mt < 4; ++mt)
      for (int nt = 0; nt < 2; ++nt)
        acc[mt][nt] = __builtin_amdgcn_wmma_f32_16x16x32_f16(
            false, Af[mt].h, false, Bf[nt].h, (short)0, acc[mt][nt], false, false);

    if (kt + 1 < nk) {
      int nxt = cur ^ 1;
      *(v4u*)&As[nxt][r0 * LDT + c0] = a0;
      *(v4u*)&As[nxt][r1 * LDT + c1] = a1;
      *(v4u*)&Bs[nxt][r0 * LDT + c0] = b0;
      *(v4u*)&Bs[nxt][r1 * LDT + c1] = b1;
    }
    __syncthreads();
  }

  // Epilogue. C layout: vgpr i -> M = i (lanes 0-15) / i+8 (lanes 16-31), N = lane%16.
  int hiOff = lo ? 0 : 8;
  for (int mt = 0; mt < 4; ++mt) {
    int rIn = waveM * 64 + mt * 16 + hiOff;
    for (int nt = 0; nt < 2; ++nt) {
      int c = colBase + waveN * 32 + nt * 16 + lr;
      float bv = bias[c];
      for (int i = 0; i < 8; ++i) {
        int r = rowBase + rIn + i;
        if (r >= M) continue;
        float v = acc[mt][nt][i] + bv;
        if (EPI == 1) {
          v = 0.5f * v * (1.0f + erff(v * 0.70710678f));
          outH[(size_t)r * N + c] = (_Float16)v;
        } else if (EPI == 2) {
          int b = r / S_, s = r % S_;
          int h = c >> 6, d = c & 63;
          outH[(((size_t)(b * NH_ + h)) * S_ + s) * DH_ + d] = (_Float16)v;
        } else if (EPI == 3) {
          int b = r / S_, s = r % S_;
          int h = c >> 6, d = c & 63;
          outH[(((size_t)(b * NH_ + h)) * DH_ + d) * S_ + s] = (_Float16)v;
        } else if (EPI == 4) {
          outF[(size_t)r * N + c] = v + res[(size_t)r * N + c];
        }
      }
    }
  }
}

// ---------------------------------------------------------------------------
// Flash attention: grid = (S/128, B*NH), 256 threads = 8 waves.
// Each wave owns 16 q-rows, streams keys 32 at a time. Softmax statistics are
// reduced with DPP; P round-trips a wave-private LDS slab (per-wave DS ops are
// in-order in hardware, so no block barrier is needed). Next K/V block is
// prefetched into L2 while the current block computes.
// q,k layout [B,NH,S,DH]; v layout [B,NH,DH,S]; ctx out [B,S,H] f16.
// ---------------------------------------------------------------------------
__global__ __launch_bounds__(256) void attention_kernel(
    const _Float16* __restrict__ qbuf, const _Float16* __restrict__ kbuf,
    const _Float16* __restrict__ vbuf, const float* __restrict__ mask,
    _Float16* __restrict__ ctx) {
  int tid = threadIdx.x;
  int w = tid >> 5, lane = tid & 31, lr = lane & 15;
  bool lo = lane < 16;
  int byteSel = lo ? 0 : 16;
  int hiOff = lo ? 0 : 8;

  int bh = blockIdx.y;
  int b = bh / NH_;
  int h = bh % NH_;
  int qrow0 = blockIdx.x * 128 + w * 16;
  size_t headOff = (size_t)bh * S_ * DH_;

  __shared__ _Float16 Plds[8][16 * 32];  // per-wave P slab (16x32 f16)

  // Q fragments (K-dim 0..31 and 32..63), loaded once.
  AFrag Qf[2];
  {
    const char* qp = (const char*)(qbuf + headOff + (size_t)(qrow0 + lr) * DH_);
    for (int j = 0; j < 2; ++j) {
      Qf[j].u[0] = *(const v4u*)(qp + j * 64 + byteSel);
      Qf[j].u[1] = *(const v4u*)(qp + j * 64 + 32 + byteSel);
    }
  }

  v8f zero = {0.f, 0.f, 0.f, 0.f, 0.f, 0.f, 0.f, 0.f};
  v8f O[4];
  for (int t = 0; t < 4; ++t) O[t] = zero;
  float Mrun[8], Lrun[8];
  for (int i = 0; i < 8; ++i) { Mrun[i] = -1e30f; Lrun[i] = 0.f; }

  const float scale = 0.125f;  // 1/sqrt(64)
  const float* mrow = mask + (size_t)b * S_;

  for (int kb0 = 0; kb0 < S_; kb0 += 32) {
    // Prefetch next key/value block into L2.
    if (kb0 + 32 < S_) {
      __builtin_prefetch(kbuf + headOff + (size_t)(kb0 + 32 + lr) * DH_, 0, 0);
      __builtin_prefetch(vbuf + (size_t)bh * DH_ * S_ + (size_t)(lr)*S_ + kb0 + 32, 0, 0);
    }
    v8f s0 = zero, s1 = zero;
    {
      const char* kp0 = (const char*)(kbuf + headOff + (size_t)(kb0 + lr) * DH_);
      const char* kp1 = (const char*)(kbuf + headOff + (size_t)(kb0 + 16 + lr) * DH_);
      for (int j = 0; j < 2; ++j) {
        AFrag Kf;
        Kf.u[0] = *(const v4u*)(kp0 + j * 64 + byteSel);
        Kf.u[1] = *(const v4u*)(kp0 + j * 64 + 32 + byteSel);
        s0 = __builtin_amdgcn_wmma_f32_16x16x32_f16(false, Qf[j].h, false, Kf.h,
                                                    (short)0, s0, false, false);
        Kf.u[0] = *(const v4u*)(kp1 + j * 64 + byteSel);
        Kf.u[1] = *(const v4u*)(kp1 + j * 64 + 32 + byteSel);
        s1 = __builtin_amdgcn_wmma_f32_16x16x32_f16(false, Qf[j].h, false, Kf.h,
                                                    (short)0, s1, false, false);
      }
    }
    float mb0 = (1.0f - mrow[kb0 + lr]) * -10000.0f;
    float mb1 = (1.0f - mrow[kb0 + 16 + lr]) * -10000.0f;

    float alpha[8];
    for (int i = 0; i < 8; ++i) {
      s0[i] = s0[i] * scale + mb0;
      s1[i] = s1[i] * scale + mb1;
      float tm = rmax16(fmaxf(s0[i], s1[i]));
      float mnew = fmaxf(Mrun[i], tm);
      alpha[i] = __expf(Mrun[i] - mnew);
      Mrun[i] = mnew;
      s0[i] = __expf(s0[i] - mnew);
      s1[i] = __expf(s1[i] - mnew);
      float rsum = radd16(s0[i] + s1[i]);
      Lrun[i] = Lrun[i] * alpha[i] + rsum;
    }
    for (int t = 0; t < 4; ++t)
      for (int i = 0; i < 8; ++i) O[t][i] *= alpha[i];

    // C-layout -> wave-private LDS slab -> A-fragment. Per-wave DS ops are
    // in-order in hardware; the asm fences stop compiler reordering.
    asm volatile("" ::: "memory");
    for (int i = 0; i < 8; ++i) {
      Plds[w][(i + hiOff) * 32 + lr] = (_Float16)s0[i];
      Plds[w][(i + hiOff) * 32 + 16 + lr] = (_Float16)s1[i];
    }
    asm volatile("" ::: "memory");
    AFrag Pf;
    {
      const char* pp = (const char*)(&Plds[w][lr * 32]);
      Pf.u[0] = *(const v4u*)(pp + byteSel);
      Pf.u[1] = *(const v4u*)(pp + 32 + byteSel);
    }
    asm volatile("" ::: "memory");
    for (int t = 0; t < 4; ++t) {
      const char* vp =
          (const char*)(vbuf + (size_t)bh * DH_ * S_ + (size_t)(t * 16 + lr) * S_ + kb0);
      AFrag Vf;
      Vf.u[0] = *(const v4u*)(vp + byteSel);
      Vf.u[1] = *(const v4u*)(vp + 32 + byteSel);
      O[t] = __builtin_amdgcn_wmma_f32_16x16x32_f16(false, Pf.h, false, Vf.h,
                                                    (short)0, O[t], false, false);
    }
  }

  for (int i = 0; i < 8; ++i) {
    float inv = 1.0f / Lrun[i];
    int q = qrow0 + i + hiOff;
    _Float16* dst = ctx + ((size_t)(b * S_ + q)) * H_ + h * DH_;
    for (int t = 0; t < 4; ++t) dst[t * 16 + lr] = (_Float16)(O[t][i] * inv);
  }
}

// ---------------------------------------------------------------------------
// Row LayerNorm over H=768; one block per row; optional f16 mirror.
// ---------------------------------------------------------------------------
__global__ __launch_bounds__(256) void layernorm_kernel(
    const float* __restrict__ x, const float* __restrict__ g,
    const float* __restrict__ bb, float* __restrict__ outF,
    _Float16* __restrict__ outH) {
  __shared__ float red[256];
  int row = blockIdx.x, tid = threadIdx.x;
  const float* xr = x + (size_t)row * H_;
  float v0 = xr[tid], v1 = xr[tid + 256], v2 = xr[tid + 512];
  red[tid] = v0 + v1 + v2;
  __syncthreads();
  for (int o = 128; o > 0; o >>= 1) {
    if (tid < o) red[tid] += red[tid + o];
    __syncthreads();
  }
  float mu = red[0] * (1.0f / 768.0f);
  __syncthreads();
  float d0 = v0 - mu, d1 = v1 - mu, d2 = v2 - mu;
  red[tid] = d0 * d0 + d1 * d1 + d2 * d2;
  __syncthreads();
  for (int o = 128; o > 0; o >>= 1) {
    if (tid < o) red[tid] += red[tid + o];
    __syncthreads();
  }
  float rs = rsqrtf(red[0] * (1.0f / 768.0f) + 1e-12f);
  float* orow = outF + (size_t)row * H_;
  float y0 = d0 * rs * g[tid] + bb[tid];
  float y1 = d1 * rs * g[tid + 256] + bb[tid + 256];
  float y2 = d2 * rs * g[tid + 512] + bb[tid + 512];
  orow[tid] = y0; orow[tid + 256] = y1; orow[tid + 512] = y2;
  if (outH) {
    _Float16* hrow = outH + (size_t)row * H_;
    hrow[tid] = (_Float16)y0;
    hrow[tid + 256] = (_Float16)y1;
    hrow[tid + 512] = (_Float16)y2;
  }
}

// ---------------------------------------------------------------------------
// Pooler: cls[b][j] = tanh(out[b,0,:] . Wc[:,j] + bc[j])
// ---------------------------------------------------------------------------
__global__ __launch_bounds__(256) void cls_kernel(const float* __restrict__ out,
                                                  const float* __restrict__ Wc,
                                                  const float* __restrict__ bc,
                                                  float* __restrict__ cls) {
  int idx = blockIdx.x * 256 + threadIdx.x;
  if (idx >= B_ * H_) return;
  int b = idx / H_, j = idx % H_;
  const float* xr = out + (size_t)b * S_ * H_;
  float acc = bc[j];
  for (int k = 0; k < H_; ++k) acc += xr[k] * Wc[(size_t)k * H_ + j];
  cls[idx] = tanhf(acc);
}

__global__ __launch_bounds__(64) void logits_kernel(const float* __restrict__ cls,
                                                    const float* __restrict__ Wp,
                                                    const float* __restrict__ bp,
                                                    float* __restrict__ logits) {
  int idx = threadIdx.x;
  if (idx >= B_ * 3) return;
  int b = idx / 3, p = idx % 3;
  float acc = bp[p];
  for (int k = 0; k < H_; ++k) acc += cls[b * H_ + k] * Wp[k * 3 + p];
  logits[idx] = acc;
}

// ---------------------------------------------------------------------------
extern "C" void kernel_launch(void* const* d_in, const int* in_sizes, int n_in,
                              void* d_out, int out_size, void* d_ws, size_t ws_size,
                              hipStream_t stream) {
  (void)in_sizes; (void)n_in; (void)out_size; (void)ws_size;
  const float* hidden = (const float*)d_in[0];
  const float* amask  = (const float*)d_in[1];
  const float* Wq = (const float*)d_in[2];  const float* bq = (const float*)d_in[3];
  const float* Wk = (const float*)d_in[4];  const float* bk = (const float*)d_in[5];
  const float* Wv = (const float*)d_in[6];  const float* bv = (const float*)d_in[7];
  const float* Wo = (const float*)d_in[8];  const float* bo = (const float*)d_in[9];
  const float* ln1g = (const float*)d_in[10]; const float* ln1b = (const float*)d_in[11];
  const float* W1 = (const float*)d_in[12]; const float* b1 = (const float*)d_in[13];
  const float* W2 = (const float*)d_in[14]; const float* b2 = (const float*)d_in[15];
  const float* ln2g = (const float*)d_in[16]; const float* ln2b = (const float*)d_in[17];
  const float* Wc = (const float*)d_in[18]; const float* bc = (const float*)d_in[19];
  const float* Wp = (const float*)d_in[20]; const float* bp = (const float*)d_in[21];
  float* out = (float*)d_out;

  const size_t MB = (size_t)B_ * S_;  // 16384 token rows

  char* ws = (char*)d_ws;
  size_t off = 0;
  auto alloc = [&](size_t bytes) -> char* {
    char* p = ws + off;
    off += (bytes + 255) & ~(size_t)255;
    return p;
  };

  _Float16* Xh  = (_Float16*)alloc(MB * H_ * 2);
  _Float16* WqT = (_Float16*)alloc((size_t)H_ * H_ * 2);
  _Float16* WkT = (_Float16*)alloc((size_t)H_ * H_ * 2);
  _Float16* WvT = (_Float16*)alloc((size_t)H_ * H_ * 2);
  _Float16* WoT = (_Float16*)alloc((size_t)H_ * H_ * 2);
  _Float16* W1T = (_Float16*)alloc((size_t)H_ * FF_ * 2);
  _Float16* W2T = (_Float16*)alloc((size_t)H_ * FF_ * 2);
  // q,k,vt,ctx are contiguous (each MB*H*2, already 256B-multiples) and are
  // exactly reused as the MB*FF*2 FFN1 activation buffer after attention.
  _Float16* qb   = (_Float16*)alloc(MB * H_ * 2);
  _Float16* kbuf = (_Float16*)alloc(MB * H_ * 2);
  _Float16* vtb  = (_Float16*)alloc(MB * H_ * 2);
  _Float16* ctx  = (_Float16*)alloc(MB * H_ * 2);
  _Float16* ffn1 = qb;  // alias: 4 * MB*H*2 == MB*FF*2
  float* pre   = (float*)alloc(MB * H_ * 4);   // GEMM+residual scratch (reused)
  float* attnF = (float*)alloc(MB * H_ * 4);   // attn_out f32
  _Float16* attnH = (_Float16*)alloc(MB * H_ * 2);
  float* clsb  = (float*)alloc((size_t)B_ * H_ * 4);

  dim3 blk256(256);

  // 1. hidden -> f16
  cast_f16_kernel<<<dim3((unsigned)((MB * H_ + 255) / 256)), blk256, 0, stream>>>(
      hidden, Xh, (int)(MB * H_));

  // 2. weight transposes to [N][K] f16
  dim3 tblk(32, 8);
  transpose_f16_kernel<<<dim3(H_ / 32, H_ / 32), tblk, 0, stream>>>(Wq, WqT, H_, H_);
  transpose_f16_kernel<<<dim3(H_ / 32, H_ / 32), tblk, 0, stream>>>(Wk, WkT, H_, H_);
  transpose_f16_kernel<<<dim3(H_ / 32, H_ / 32), tblk, 0, stream>>>(Wv, WvT, H_, H_);
  transpose_f16_kernel<<<dim3(H_ / 32, H_ / 32), tblk, 0, stream>>>(Wo, WoT, H_, H_);
  transpose_f16_kernel<<<dim3(FF_ / 32, H_ / 32), tblk, 0, stream>>>(W1, W1T, H_, FF_);
  transpose_f16_kernel<<<dim3(H_ / 32, FF_ / 32), tblk, 0, stream>>>(W2, W2T, FF_, H_);

  // 3-5. QKV projections (WMMA), writing head layouts directly
  dim3 gHH(H_ / 128, (unsigned)(MB / 128));
  gemm_wmma_kernel<2><<<gHH, blk256, 0, stream>>>(Xh, WqT, bq, (int)MB, H_, H_,
                                                  qb, nullptr, nullptr);
  gemm_wmma_kernel<2><<<gHH, blk256, 0, stream>>>(Xh, WkT, bk, (int)MB, H_, H_,
                                                  kbuf, nullptr, nullptr);
  gemm_wmma_kernel<3><<<gHH, blk256, 0, stream>>>(Xh, WvT, bv, (int)MB, H_, H_,
                                                  vtb, nullptr, nullptr);

  // 6. flash attention (WMMA QK^T and PV)
  attention_kernel<<<dim3(S_ / 128, B_ * NH_), blk256, 0, stream>>>(qb, kbuf, vtb,
                                                                    amask, ctx);

  // 7. O projection + bias + residual(hidden) -> f32
  gemm_wmma_kernel<4><<<gHH, blk256, 0, stream>>>(ctx, WoT, bo, (int)MB, H_, H_,
                                                  nullptr, pre, hidden);

  // 8. LN1 -> attn_out (f32 + f16 mirror)
  layernorm_kernel<<<dim3((unsigned)MB), blk256, 0, stream>>>(pre, ln1g, ln1b,
                                                              attnF, attnH);

  // 9. FFN1 with exact GELU -> f16 (reuses q/k/vt/ctx region)
  dim3 gFF(FF_ / 128, (unsigned)(MB / 128));
  gemm_wmma_kernel<1><<<gFF, blk256, 0, stream>>>(attnH, W1T, b1, (int)MB, FF_, H_,
                                                  ffn1, nullptr, nullptr);

  // 10. FFN2 + bias + residual(attn_out) -> f32 (pre reused)
  gemm_wmma_kernel<4><<<gHH, blk256, 0, stream>>>(ffn1, W2T, b2, (int)MB, H_, FF_,
                                                  nullptr, pre, attnF);

  // 11. LN2 straight into d_out
  layernorm_kernel<<<dim3((unsigned)MB), blk256, 0, stream>>>(pre, ln2g, ln2b,
                                                              out, nullptr);

  // 12-13. pooler + logits
  cls_kernel<<<dim3((B_ * H_ + 255) / 256), blk256, 0, stream>>>(out, Wc, bc, clsb);
  logits_kernel<<<dim3(1), dim3(64), 0, stream>>>(clsb, Wp, bp, out + MB * H_);
}
